// AttentionHead_63677185130745
// MI455X (gfx1250) — compile-verified, matrix operand order checked by reading, not run on hardware
//
#include <hip/hip_runtime.h>
#include <hip/hip_bf16.h>

// ---------------------------------------------------------------------------
// Types for CDNA5 WMMA (wave32): v_wmma_f32_16x16x32_bf16
// ---------------------------------------------------------------------------
typedef __bf16 bf16_t;
typedef bf16_t v16bf __attribute__((ext_vector_type(16)));
typedef bf16_t v8bf  __attribute__((ext_vector_type(8)));
typedef float  v8f   __attribute__((ext_vector_type(8)));

#define D_MODEL 1024
#define S_LEN   2048
#define BATCH   4
#define INV_SCALE 0.03125f   // 1/sqrt(1024)

#define NEG_INF (-__builtin_inff())

// ---------------------------------------------------------------------------
// Fragment loader: 16-bit A (16x32) / B (32x16, from row-major Bt = B^T).
// ISA 7.12.2: lane holds row/col = lane&15; half = lane>>4 selects K groups:
//   frag[0..7]  = mem[k + 8*half + 0..7]
//   frag[8..15] = mem[k + 16 + 8*half + 0..7]
// Both runs are contiguous 16-byte loads (global_load_b128 / ds_load_b128).
// ---------------------------------------------------------------------------
template <typename PtrT>
__device__ __forceinline__ v16bf load_frag(PtrT p, int half) {
    const v8bf lo = *reinterpret_cast<const v8bf*>(p + 8 * half);
    const v8bf hi = *reinterpret_cast<const v8bf*>(p + 16 + 8 * half);
    v16bf f;
#pragma unroll
    for (int e = 0; e < 8; ++e) { f[e] = lo[e]; f[8 + e] = hi[e]; }
    return f;
}

__device__ __forceinline__ v8f wmma_bf16(v16bf a, v16bf b, v8f c) {
    return __builtin_amdgcn_wmma_f32_16x16x32_bf16(
        /*neg_a=*/false, a, /*neg_b=*/false, b,
        /*c_mod=*/(short)0, c, /*reuse_a=*/false, /*reuse_b=*/false);
}

// ---------------------------------------------------------------------------
// Converters
// ---------------------------------------------------------------------------
__global__ void cvt_x_kernel(const float* __restrict__ x,
                             bf16_t* __restrict__ xb,   // [B*S][D]
                             bf16_t* __restrict__ xt) { // [B][D][S]
    size_t idx = (size_t)blockIdx.x * blockDim.x + threadIdx.x;
    int d = (int)(idx % D_MODEL);
    size_t bs = idx / D_MODEL;
    int s = (int)(bs % S_LEN);
    int b = (int)(bs / S_LEN);
    bf16_t h = (bf16_t)x[idx];
    xb[idx] = h;
    xt[((size_t)b * D_MODEL + d) * S_LEN + s] = h;
}

__global__ void cvt_wT_kernel(const float* __restrict__ w,   // [K][N]
                              bf16_t* __restrict__ wt) {     // [N][K]
    size_t idx = (size_t)blockIdx.x * blockDim.x + threadIdx.x;
    int n = (int)(idx % D_MODEL);
    int k = (int)(idx / D_MODEL);
    wt[(size_t)n * D_MODEL + k] = (bf16_t)w[idx];
}

// ---------------------------------------------------------------------------
// GEMM: C[M][N] = A[M][K] * B[K][N], with B given transposed (Bt[N][K]).
// Block = 8 waves; wave owns a 16x128 tile (8 accumulators, 8x A-frag reuse);
// grid (N/1024, M/16).
// ---------------------------------------------------------------------------
template <typename OutT>
__global__ __launch_bounds__(256)
void gemm_bt_kernel(const bf16_t* __restrict__ A,
                    const bf16_t* __restrict__ Bt,
                    OutT* __restrict__ C,
                    int M, int N, int K) {
    const int lane = threadIdx.x & 31;
    const int wave = threadIdx.x >> 5;
    const int row  = lane & 15;
    const int half = lane >> 4;
    const int m0 = blockIdx.y * 16;
    const int n0 = blockIdx.x * 1024 + wave * 128;

    v8f acc[8];
#pragma unroll
    for (int t = 0; t < 8; ++t)
#pragma unroll
        for (int r = 0; r < 8; ++r) acc[t][r] = 0.0f;

    const bf16_t* arow = A + (size_t)(m0 + row) * K;
    for (int k = 0; k < K; k += 32) {
        __builtin_prefetch(arow + k + 128, 0, 0);
        const v16bf af = load_frag(arow + k, half);
#pragma unroll
        for (int t = 0; t < 8; ++t) {
            const bf16_t* brow = Bt + (size_t)(n0 + t * 16 + row) * K + k;
            const v16bf bfr = load_frag(brow, half);
            acc[t] = wmma_bf16(af, bfr, acc[t]);
        }
    }
#pragma unroll
    for (int t = 0; t < 8; ++t)
#pragma unroll
        for (int r = 0; r < 8; ++r) {
            const int gm = m0 + r + 8 * half;
            const int gn = n0 + t * 16 + row;
            C[(size_t)gm * N + gn] = (OutT)acc[t][r];
        }
}

// ---------------------------------------------------------------------------
// Flash-attention core.
// Workgroup: 16 query rows (m0..m0+15), 16 waves (512 threads).
// Key chunks of KC=512: scores staged as f32 in LDS, softmax writes bf16
// probabilities to a second LDS buffer so phase-3 A-fragments are pure
// ds_load_b128 (no f32->bf16 VALU in the P*V hot loop). Each wave owns a
// 16x64 slice of O1 = softmax(Q K^T / sqrt(D)) * V.
// ---------------------------------------------------------------------------
#define KC 512
#define SCPAD 520   // f32 score row stride (de-phase banks)
#define PBPAD 528   // bf16 prob row stride (16B multiple, de-phased)

__global__ __launch_bounds__(512)
void attn_core_kernel(const bf16_t* __restrict__ Q,   // [B*S][D]
                      const bf16_t* __restrict__ Kx,  // [B*S][D]  (= x)
                      const bf16_t* __restrict__ Vt,  // [B][D][S] (= x^T)
                      bf16_t* __restrict__ O1) {      // [B*S][D]
    __shared__ __attribute__((aligned(16))) float  sc[16][SCPAD];
    __shared__ __attribute__((aligned(16))) bf16_t pb[16][PBPAD];
    __shared__ float s_run_max[16];
    __shared__ float s_run_sum[16];
    __shared__ float s_scale[16];

    const int lane = threadIdx.x & 31;
    const int wave = threadIdx.x >> 5;
    const int row  = lane & 15;
    const int half = lane >> 4;

    const int blocksPerBatch = S_LEN / 16;
    const int b  = blockIdx.x / blocksPerBatch;
    const int m0 = (blockIdx.x % blocksPerBatch) * 16;

    const bf16_t* Qb = Q  + (size_t)b * S_LEN * D_MODEL;
    const bf16_t* Kb = Kx + (size_t)b * S_LEN * D_MODEL;
    const bf16_t* Vb = Vt + (size_t)b * D_MODEL * S_LEN;

    if (threadIdx.x < 16) { s_run_max[threadIdx.x] = NEG_INF; s_run_sum[threadIdx.x] = 0.0f; }

    v8f acc[4];
#pragma unroll
    for (int t = 0; t < 4; ++t)
#pragma unroll
        for (int r = 0; r < 8; ++r) acc[t][r] = 0.0f;

    const int kend = m0 + 16;                 // causal: keys 0..m0+15 needed
    const int nchunks = (kend + KC - 1) / KC;

    for (int c = 0; c < nchunks; ++c) {
        const int k0c = c * KC;
        __syncthreads();                      // LDS reuse fence (also covers init)

        // ---- Phase 1: score tiles for keys [k0c, k0c+KC) --------------------
        for (int t = wave; t < KC / 16; t += 16) {
            const int n0 = k0c + t * 16;      // global key tile start (uniform)
            v8f ct;
#pragma unroll
            for (int r = 0; r < 8; ++r) ct[r] = 0.0f;
            const bool live = (n0 <= m0 + 15);   // wave-uniform -> EXEC stays full
            if (live) {
                const bf16_t* qrow = Qb + (size_t)(m0 + row) * D_MODEL;
                const bf16_t* krow = Kb + (size_t)(n0 + row) * D_MODEL;
                for (int kk = 0; kk < D_MODEL; kk += 32) {
                    const v16bf af  = load_frag(qrow + kk, half);
                    const v16bf bfr = load_frag(krow + kk, half);
                    ct = wmma_bf16(af, bfr, ct);
                }
            }
            const int gn = n0 + row;          // C-tile column = key index
#pragma unroll
            for (int r = 0; r < 8; ++r) {
                const int gm = m0 + r + 8 * half;
                float v = live ? ct[r] * INV_SCALE : NEG_INF;
                if (gn > gm) v = NEG_INF;     // causal mask
                sc[r + 8 * half][t * 16 + row] = v;
            }
        }
        __syncthreads();

        // ---- Phase 2: online softmax update; wave w owns row w --------------
        {
            const float* srow = sc[wave];
            bf16_t* prow = pb[wave];
            float mx = NEG_INF;
            for (int i = lane; i < KC; i += 32) mx = fmaxf(mx, srow[i]);
#pragma unroll
            for (int off = 16; off > 0; off >>= 1) mx = fmaxf(mx, __shfl_xor(mx, off, 32));
            const float oldm = s_run_max[wave];
            const float newm = fmaxf(oldm, mx);
            float sum = 0.0f;
            for (int i = lane; i < KC; i += 32) {
                const float p = __expf(srow[i] - newm);
                prow[i] = (bf16_t)p;          // bf16 probs for the WMMA A-matrix
                sum += p;                     // f32 sum for exact normalization
            }
#pragma unroll
            for (int off = 16; off > 0; off >>= 1) sum += __shfl_xor(sum, off, 32);
            if (lane == 0) {
                const float rs = __expf(oldm - newm);
                s_scale[wave]   = rs;
                s_run_sum[wave] = s_run_sum[wave] * rs + sum;
                s_run_max[wave] = newm;
            }
        }
        __syncthreads();

        // ---- Phase 3: rescale accumulators, O1 += P * V ---------------------
#pragma unroll
        for (int t = 0; t < 4; ++t)
#pragma unroll
            for (int r = 0; r < 8; ++r) acc[t][r] *= s_scale[r + 8 * half];

        int klen = kend - k0c; if (klen > KC) klen = KC;
        const int klen32 = (klen + 31) & ~31;        // padded cols hold P==0
        for (int kk = 0; kk < klen32; kk += 32) {
            const v16bf af = load_frag(&pb[row][kk], half);   // ds_load_b128 x2
#pragma unroll
            for (int t = 0; t < 4; ++t) {
                const int dcol = wave * 64 + t * 16 + row;    // B-frag col = out dim
                const bf16_t* vrow = Vb + (size_t)dcol * S_LEN + k0c + kk;
                const v16bf bfr = load_frag(vrow, half);
                acc[t] = wmma_bf16(af, bfr, acc[t]);
            }
        }
    }

    // ---- Epilogue: normalize and store O1 (bf16) ---------------------------
    bf16_t* ob = O1 + (size_t)b * S_LEN * D_MODEL;
#pragma unroll
    for (int t = 0; t < 4; ++t)
#pragma unroll
        for (int r = 0; r < 8; ++r) {
            const int gm = m0 + r + 8 * half;
            const int dn = wave * 64 + t * 16 + row;
            const float denom = s_run_sum[r + 8 * half];
            ob[(size_t)gm * D_MODEL + dn] = (bf16_t)(acc[t][r] / denom);
        }
}

// ---------------------------------------------------------------------------
// Host launcher
// ---------------------------------------------------------------------------
extern "C" void kernel_launch(void* const* d_in, const int* in_sizes, int n_in,
                              void* d_out, int out_size, void* d_ws, size_t ws_size,
                              hipStream_t stream) {
    (void)in_sizes; (void)n_in; (void)out_size; (void)ws_size;
    const float* x  = (const float*)d_in[0];   // [B,S,D]
    const float* qk = (const float*)d_in[1];   // [D,D]
    const float* ov = (const float*)d_in[2];   // [D,D]
    float* out = (float*)d_out;                // [B,S,D] f32

    const size_t NX = (size_t)BATCH * S_LEN * D_MODEL;   // 8,388,608
    const size_t NW = (size_t)D_MODEL * D_MODEL;         // 1,048,576

    char* ws = (char*)d_ws;
    bf16_t* xb   = (bf16_t*)(ws);                         // 16 MB  x row-major
    bf16_t* xt   = (bf16_t*)(ws + NX * 2);                // 16 MB  x transposed
    bf16_t* qkT  = (bf16_t*)(ws + NX * 4);                //  2 MB
    bf16_t* ovT  = (bf16_t*)(ws + NX * 4 + NW * 2);       //  2 MB
    bf16_t* Qm   = (bf16_t*)(ws + NX * 4 + NW * 4);       // 16 MB  Q = x@qk
    bf16_t* O1   = (bf16_t*)(ws + NX * 6 + NW * 4);       // 16 MB  attn@x

    // 1) precision conversion (+ transposes)
    cvt_x_kernel<<<dim3((unsigned)(NX / 256)), dim3(256), 0, stream>>>(x, xb, xt);
    cvt_wT_kernel<<<dim3((unsigned)(NW / 256)), dim3(256), 0, stream>>>(qk, qkT);
    cvt_wT_kernel<<<dim3((unsigned)(NW / 256)), dim3(256), 0, stream>>>(ov, ovT);

    // 2) Q = x @ qk   (M = B*S = 8192, N = K = 1024), bf16 output
    gemm_bt_kernel<bf16_t><<<dim3(D_MODEL / 1024, (BATCH * S_LEN) / 16), dim3(256), 0, stream>>>(
        xb, qkT, Qm, BATCH * S_LEN, D_MODEL, D_MODEL);

    // 3) O1 = softmax(Q x^T / sqrt(D), causal) @ x
    attn_core_kernel<<<dim3(BATCH * (S_LEN / 16)), dim3(512), 0, stream>>>(Qm, xb, xt, O1);

    // 4) out = O1 @ ov, f32 output
    gemm_bt_kernel<float><<<dim3(D_MODEL / 1024, (BATCH * S_LEN) / 16), dim3(256), 0, stream>>>(
        O1, ovT, out, BATCH * S_LEN, D_MODEL, D_MODEL);
}
